// Informer_32195074851275
// MI455X (gfx1250) — matrix operand order repaired and checked by reading, not compiled, not run on hardware
//
#include <hip/hip_runtime.h>
#include <hip/hip_bf16.h>
#include <math.h>

// ---------------------------------------------------------------------------
// Model dims
// ---------------------------------------------------------------------------
#define SEQ    1000
#define SPAD   1024          // padded sequence (multiple of 256 GEMM M-tile)
#define BATCH  16
#define DMODEL 512
#define DIMFF  2048
#define NFEAT  64
#define KTOP   100           // int(1000 * 0.1)

typedef __attribute__((ext_vector_type(16))) __bf16 bf16x16;
typedef __attribute__((ext_vector_type(8)))  float  f32x8;

// ---------------------------------------------------------------------------
// Batched NT-form WMMA GEMM:  C[z][M,N] = A[z][M,K] * B[z][N,K]^T
// bf16 operands, f32 accumulate.
// Block: 256 threads = 8 waves; block tile 256(M) x 64(N); wave tile 32x64
// (2 A-fragments x 4 B-fragments = 8 WMMAs per 32-wide K-step, so each LDS
// B-fragment is reused twice and DS waits are amortized).
// B tile double-buffered in LDS; next-step A/B fetches software-pipelined.
// Requires M % 256 == 0, N % 64 == 0, K % 32 == 0 (caller pads).
// ---------------------------------------------------------------------------
enum { EPI_SCALE_F32 = 0, EPI_F32 = 1, EPI_BIAS_RELU_BF16 = 2, EPI_BIAS_F32 = 3 };

template<int EPI>
__global__ __launch_bounds__(256)
void gemm_nt_wmma(const __bf16* __restrict__ A, const __bf16* __restrict__ B,
                  float* __restrict__ Cf, __bf16* __restrict__ Cb,
                  const float* __restrict__ bias,
                  int K, int lda, int ldb, int ldc, float scale,
                  size_t strideA, size_t strideB, size_t strideC)
{
  __shared__ __bf16 Bs[2][64 * 32];      // [buf][col*32 + k], 2 x 4 KB

  A += (size_t)blockIdx.z * strideA;
  B += (size_t)blockIdx.z * strideB;

  const int tid  = threadIdx.x;
  const int lane = tid & 31;
  const int wave = tid >> 5;
  const int ln   = lane & 15;            // row/col within 16
  const int lh   = lane >> 4;            // lane-half (K split)
  const int n0   = blockIdx.x * 64;
  const int mrow = blockIdx.y * 256 + wave * 32;

  // B-staging assignment: one uint4 (8 bf16) per thread per K-step
  const int scol = tid >> 2;             // 0..63
  const int skp  = (tid & 3) * 8;        // 0,8,16,24
  const __bf16* bstage_base = B + (size_t)(n0 + scol) * ldb + skp;

  f32x8 acc[2][4];
  #pragma unroll
  for (int r = 0; r < 2; ++r)
    #pragma unroll
    for (int i = 0; i < 4; ++i) acc[r][i] = (f32x8){};

  const __bf16* arow0 = A + (size_t)(mrow + ln) * lda;
  const __bf16* arow1 = A + (size_t)(mrow + 16 + ln) * lda;

  // ---- prologue: stage B(k=0) into buffer 0, load A fragments (k=0) ----
  {
    uint4 s0 = *(const uint4*)(bstage_base);
    *(uint4*)&Bs[0][scol * 32 + skp] = s0;
  }
  bf16x16 afc[2];
  {
    uint4 u0 = *(const uint4*)(arow0 + lh * 8);
    uint4 u1 = *(const uint4*)(arow0 + 16 + lh * 8);
    __builtin_memcpy(&afc[0], &u0, 16);
    __builtin_memcpy(reinterpret_cast<char*>(&afc[0]) + 16, &u1, 16);
    uint4 u2 = *(const uint4*)(arow1 + lh * 8);
    uint4 u3 = *(const uint4*)(arow1 + 16 + lh * 8);
    __builtin_memcpy(&afc[1], &u2, 16);
    __builtin_memcpy(reinterpret_cast<char*>(&afc[1]) + 16, &u3, 16);
  }
  __syncthreads();

  int buf = 0;
  for (int kk = 0; kk < K; kk += 32) {
    const bool more = (kk + 32 < K);

    // ---- start next-step global fetches (overlap with this step's WMMAs) --
    uint4 bstage = {};
    if (more) bstage = *(const uint4*)(bstage_base + kk + 32);

    bf16x16 afn[2] = { afc[0], afc[1] };
    if (more) {
      uint4 u0 = *(const uint4*)(arow0 + kk + 32 + lh * 8);
      uint4 u1 = *(const uint4*)(arow0 + kk + 48 + lh * 8);
      __builtin_memcpy(&afn[0], &u0, 16);
      __builtin_memcpy(reinterpret_cast<char*>(&afn[0]) + 16, &u1, 16);
      uint4 u2 = *(const uint4*)(arow1 + kk + 32 + lh * 8);
      uint4 u3 = *(const uint4*)(arow1 + kk + 48 + lh * 8);
      __builtin_memcpy(&afn[1], &u2, 16);
      __builtin_memcpy(reinterpret_cast<char*>(&afn[1]) + 16, &u3, 16);
      __builtin_prefetch(arow0 + kk + 64, 0, 1);   // global_prefetch_b8
      __builtin_prefetch(arow1 + kk + 64, 0, 1);
    }

    // ---- read all 4 B fragments from LDS, then 8 back-to-back WMMAs ------
    bf16x16 bf[4];
    #pragma unroll
    for (int nt = 0; nt < 4; ++nt) {
      const uint4* q = (const uint4*)&Bs[buf][(nt * 16 + ln) * 32 + lh * 8];
      uint4 v0 = q[0];
      uint4 v1 = q[2];
      __builtin_memcpy(&bf[nt], &v0, 16);
      __builtin_memcpy(reinterpret_cast<char*>(&bf[nt]) + 16, &v1, 16);
    }
    #pragma unroll
    for (int nt = 0; nt < 4; ++nt) {
      acc[0][nt] = __builtin_amdgcn_wmma_f32_16x16x32_bf16(
          false, afc[0], false, bf[nt], (short)0, acc[0][nt], false, false);
      acc[1][nt] = __builtin_amdgcn_wmma_f32_16x16x32_bf16(
          false, afc[1], false, bf[nt], (short)0, acc[1][nt], false, false);
    }

    // ---- publish next B tile into the other buffer ------------------------
    if (more) *(uint4*)&Bs[buf ^ 1][scol * 32 + skp] = bstage;
    __syncthreads();

    afc[0] = afn[0];
    afc[1] = afn[1];
    buf ^= 1;
  }

  // C layout: VGPR v, lanes 0-15 -> M = v, lanes 16-31 -> M = v + 8
  float*  Cfo = Cf ? Cf + (size_t)blockIdx.z * strideC : (float*)0;
  __bf16* Cbo = Cb ? Cb + (size_t)blockIdx.z * strideC : (__bf16*)0;
  #pragma unroll
  for (int r = 0; r < 2; ++r) {
    const int mbase = mrow + r * 16 + lh * 8;
    #pragma unroll
    for (int nt = 0; nt < 4; ++nt) {
      const int col = n0 + nt * 16 + ln;
      float bv = 0.0f;
      if (EPI == EPI_BIAS_RELU_BF16 || EPI == EPI_BIAS_F32) bv = bias[col];
      #pragma unroll
      for (int v = 0; v < 8; ++v) {
        const size_t o = (size_t)(mbase + v) * ldc + col;
        const float val = acc[r][nt][v];
        if      (EPI == EPI_SCALE_F32)      Cfo[o] = val * scale;
        else if (EPI == EPI_F32)            Cfo[o] = val;
        else if (EPI == EPI_BIAS_F32)       Cfo[o] = val + bv;
        else /* EPI_BIAS_RELU_BF16 */       Cbo[o] = (__bf16)fmaxf(val + bv, 0.0f);
      }
    }
  }
}

// ---------------------------------------------------------------------------
// Encoder: h[b,s,:] = (x[s,b,:] @ enc_w + enc_b) * sqrt(D) + PE(s,:)
// x layout [S, B, F]; writes f32 h and bf16 copy (padded rows pre-zeroed).
// ---------------------------------------------------------------------------
__global__ void encoder_kernel(const float* __restrict__ x,
                               const float* __restrict__ w,
                               const float* __restrict__ bias,
                               float* __restrict__ h, __bf16* __restrict__ hb)
{
  const size_t idx = (size_t)blockIdx.x * blockDim.x + threadIdx.x;
  if (idx >= (size_t)BATCH * SEQ * DMODEL) return;
  const int d = (int)(idx & (DMODEL - 1));
  const int t = (int)(idx >> 9);
  const int b = t / SEQ;
  const int s = t % SEQ;

  const float* xr = x + ((size_t)s * BATCH + b) * NFEAT;
  float acc = 0.0f;
  #pragma unroll
  for (int f = 0; f < NFEAT; ++f) acc += xr[f] * w[f * DMODEL + d];
  acc = (acc + bias[d]) * 22.62741699796952f;   // sqrt(512)

  const int j = d & ~1;
  const float rate = __expf(-9.210340371976184f * (float)j * (1.0f / DMODEL)); // 10000^{-j/D}
  const float ang  = (float)s * rate;
  acc += (d & 1) ? __cosf(ang) : __sinf(ang);

  const size_t o = ((size_t)b * SPAD + s) * DMODEL + d;
  h[o]  = acc;
  hb[o] = (__bf16)acc;
}

// ---------------------------------------------------------------------------
// ProbSparse top-k + softmax. One block (256 threads) per (query, batch).
// sparse[q,j] = score if score among top-k else 0; softmax over full row.
// Top-k realized via binary-searched threshold (exact up to value ties).
// ---------------------------------------------------------------------------
__global__ __launch_bounds__(256)
void topk_softmax_kernel(const float* __restrict__ scores,
                         __bf16* __restrict__ attn)
{
  __shared__ float sc[SPAD];
  __shared__ float redf[256];
  __shared__ int   redi[256];
  const int tid = threadIdx.x;
  const size_t rowidx = (size_t)blockIdx.y * SPAD + blockIdx.x;  // [b][q]
  const float* row = scores + rowidx * SPAD;

  for (int i = tid; i < SPAD; i += 256) sc[i] = (i < SEQ) ? row[i] : -3.0e38f;
  __syncthreads();

  // row max / min
  float mx = -3.0e38f, mn = 3.0e38f;
  for (int i = tid; i < SEQ; i += 256) { float v = sc[i]; mx = fmaxf(mx, v); mn = fminf(mn, v); }
  redf[tid] = mx; __syncthreads();
  for (int st = 128; st > 0; st >>= 1) { if (tid < st) redf[tid] = fmaxf(redf[tid], redf[tid + st]); __syncthreads(); }
  const float maxv = redf[0]; __syncthreads();
  redf[tid] = mn; __syncthreads();
  for (int st = 128; st > 0; st >>= 1) { if (tid < st) redf[tid] = fminf(redf[tid], redf[tid + st]); __syncthreads(); }
  const float minv = redf[0]; __syncthreads();

  // binary search threshold t s.t. count(v >= t) ~= KTOP
  float lo = minv, hi = maxv;
  for (int it = 0; it < 24; ++it) {
    const float mid = 0.5f * (lo + hi);
    int c = 0;
    for (int i = tid; i < SEQ; i += 256) c += (sc[i] >= mid) ? 1 : 0;
    redi[tid] = c; __syncthreads();
    for (int st = 128; st > 0; st >>= 1) { if (tid < st) redi[tid] += redi[tid + st]; __syncthreads(); }
    const int cnt = redi[0]; __syncthreads();
    if (cnt >= KTOP) lo = mid; else hi = mid;
  }
  const float thresh = lo;

  // softmax over {selected scores, zeros elsewhere}
  const float m = fmaxf(maxv, 0.0f);
  int   c = 0;
  float dsum = 0.0f;
  for (int i = tid; i < SEQ; i += 256) {
    const float v = sc[i];
    if (v >= thresh) { ++c; dsum += __expf(v - m); }
  }
  redi[tid] = c; redf[tid] = dsum; __syncthreads();
  for (int st = 128; st > 0; st >>= 1) {
    if (tid < st) { redi[tid] += redi[tid + st]; redf[tid] += redf[tid + st]; }
    __syncthreads();
  }
  const int   cnt    = redi[0];
  const float sumsel = redf[0];
  __syncthreads();

  const float denom = sumsel + (float)(SEQ - cnt) * __expf(-m);
  const float inv   = 1.0f / denom;
  const float base  = __expf(-m) * inv;

  __bf16* arow = attn + rowidx * SPAD;
  for (int i = tid; i < SPAD; i += 256) {
    float w = 0.0f;
    if (i < SEQ) {
      const float v = sc[i];
      w = (v >= thresh) ? __expf(v - m) * inv : base;
    }
    arow[i] = (__bf16)w;
  }
}

// ---------------------------------------------------------------------------
// Residual add + LayerNorm over D=512. One block per live token (b,s).
// Updates h (f32) and hb (bf16).
// ---------------------------------------------------------------------------
__global__ __launch_bounds__(256)
void add_ln_kernel(float* __restrict__ h, const float* __restrict__ res,
                   const float* __restrict__ g, const float* __restrict__ be,
                   __bf16* __restrict__ hb)
{
  const int t = blockIdx.x;            // 0 .. B*SEQ-1
  const int b = t / SEQ, s = t % SEQ;
  const size_t base = ((size_t)b * SPAD + s) * DMODEL;
  const int tid = threadIdx.x;

  const float v0 = h[base + tid]       + res[base + tid];
  const float v1 = h[base + tid + 256] + res[base + tid + 256];

  __shared__ float red[256];
  red[tid] = v0 + v1; __syncthreads();
  for (int st = 128; st > 0; st >>= 1) { if (tid < st) red[tid] += red[tid + st]; __syncthreads(); }
  const float mean = red[0] * (1.0f / DMODEL); __syncthreads();

  const float d0 = v0 - mean, d1 = v1 - mean;
  red[tid] = d0 * d0 + d1 * d1; __syncthreads();
  for (int st = 128; st > 0; st >>= 1) { if (tid < st) red[tid] += red[tid + st]; __syncthreads(); }
  const float var = red[0] * (1.0f / DMODEL);

  const float r  = rsqrtf(var + 1e-5f);
  const float y0 = d0 * r * g[tid]       + be[tid];
  const float y1 = d1 * r * g[tid + 256] + be[tid + 256];
  h[base + tid]        = y0;  hb[base + tid]        = (__bf16)y0;
  h[base + tid + 256]  = y1;  hb[base + tid + 256]  = (__bf16)y1;
}

// ---------------------------------------------------------------------------
// Transposes (all write full padded extents; inputs already zero-padded)
// ---------------------------------------------------------------------------
__global__ void transpose_h_kernel(const __bf16* __restrict__ hb, __bf16* __restrict__ hT)
{
  // over BATCH * DMODEL * SPAD outputs; hT[b][d][s] = hb[b][s][d]
  const size_t o  = (size_t)blockIdx.x * blockDim.x + threadIdx.x;
  const int    b  = (int)(o / ((size_t)DMODEL * SPAD));
  const size_t r  = o % ((size_t)DMODEL * SPAD);
  const int    d  = (int)(r / SPAD), s = (int)(r % SPAD);
  hT[o] = hb[((size_t)b * SPAD + s) * DMODEL + d];
}

__global__ void transpose_w1_kernel(const float* __restrict__ w1, __bf16* __restrict__ w1t)
{
  const size_t o = (size_t)blockIdx.x * blockDim.x + threadIdx.x; // over 2048*512
  const int n = (int)(o / DMODEL), k = (int)(o % DMODEL);
  w1t[o] = (__bf16)w1[(size_t)k * DIMFF + n];
}

__global__ void transpose_w2_kernel(const float* __restrict__ w2, __bf16* __restrict__ w2t)
{
  const size_t o = (size_t)blockIdx.x * blockDim.x + threadIdx.x; // over 512*2048
  const int n = (int)(o / DIMFF), k = (int)(o % DIMFF);
  w2t[o] = (__bf16)w2[(size_t)k * DMODEL + n];
}

// ---------------------------------------------------------------------------
// Decoder: out[b,c] = h[b, SEQ-1, :] @ dec_w[:,c] + dec_b[c]   (16x2)
// ---------------------------------------------------------------------------
__global__ void decoder_kernel(const float* __restrict__ h,
                               const float* __restrict__ dw,
                               const float* __restrict__ db,
                               float* __restrict__ out)
{
  const int i = threadIdx.x;           // 32 threads = 1 wave
  const int b = i >> 1, c = i & 1;
  const float* hr = h + ((size_t)b * SPAD + (SEQ - 1)) * DMODEL;
  float acc = db[c];
  for (int d = 0; d < DMODEL; ++d) acc += hr[d] * dw[d * 2 + c];
  out[b * 2 + c] = acc;
}

// ---------------------------------------------------------------------------
// Host-side orchestration
// ---------------------------------------------------------------------------
extern "C" void kernel_launch(void* const* d_in, const int* in_sizes, int n_in,
                              void* d_out, int out_size, void* d_ws, size_t ws_size,
                              hipStream_t stream)
{
  (void)in_sizes; (void)n_in; (void)out_size; (void)ws_size;
  const float* x     = (const float*)d_in[0];
  const float* enc_w = (const float*)d_in[1];
  const float* enc_b = (const float*)d_in[2];
  const float* l1_w  = (const float*)d_in[3];
  const float* l1_b  = (const float*)d_in[4];
  const float* l2_w  = (const float*)d_in[5];
  const float* l2_b  = (const float*)d_in[6];
  const float* n1_s  = (const float*)d_in[7];
  const float* n1_b  = (const float*)d_in[8];
  const float* n2_s  = (const float*)d_in[9];
  const float* n2_b  = (const float*)d_in[10];
  const float* dec_w = (const float*)d_in[11];
  const float* dec_b = (const float*)d_in[12];
  float* out = (float*)d_out;

  // Workspace carve-up (~268 MB)
  char* p = (char*)d_ws;
  auto carve = [&](size_t bytes) -> void* {
    void* r = (void*)p;
    p += (bytes + 255) & ~(size_t)255;
    return r;
  };
  const size_t hStride  = (size_t)SPAD * DMODEL;     // per-batch token-major
  const size_t sStride  = (size_t)SPAD * SPAD;       // per-batch score matrix
  const size_t tStride  = (size_t)DMODEL * SPAD;     // per-batch h-transpose

  float*  h      = (float*) carve(BATCH * hStride * 4);
  float*  tmp    = (float*) carve(BATCH * hStride * 4);
  __bf16* hb     = (__bf16*)carve(BATCH * hStride * 2);
  __bf16* mid    = (__bf16*)carve((size_t)BATCH * SPAD * DIMFF * 2);
  __bf16* hT     = (__bf16*)carve(BATCH * tStride * 2);
  float*  scores = (float*) carve(BATCH * sStride * 4);
  __bf16* attn   = (__bf16*)carve(BATCH * sStride * 2);
  __bf16* w1t    = (__bf16*)carve((size_t)DIMFF * DMODEL * 2);
  __bf16* w2t    = (__bf16*)carve((size_t)DMODEL * DIMFF * 2);

  // Zero pads once per launch (deterministic, graph-capture safe)
  hipMemsetAsync(hb,   0, BATCH * hStride * 2, stream);
  hipMemsetAsync(attn, 0, BATCH * sStride * 2, stream);

  // Encode + positional encoding
  {
    const size_t total = (size_t)BATCH * SEQ * DMODEL;
    encoder_kernel<<<(unsigned)((total + 255) / 256), 256, 0, stream>>>(x, enc_w, enc_b, h, hb);
  }

  const float sscale = 0.04419417382415922f; // 1/sqrt(512)

  for (int l = 0; l < 3; ++l) {
    transpose_w1_kernel<<<(DIMFF * DMODEL) / 256, 256, 0, stream>>>(l1_w + (size_t)l * DMODEL * DIMFF, w1t);
    transpose_w2_kernel<<<(DMODEL * DIMFF) / 256, 256, 0, stream>>>(l2_w + (size_t)l * DIMFF * DMODEL, w2t);

    // ---- ProbSparse self-attention, all batches via blockIdx.z ----
    transpose_h_kernel<<<(unsigned)((BATCH * tStride) / 256), 256, 0, stream>>>(hb, hT);

    // scores = (h_b · h_bᵀ) / sqrt(D)   per batch [1024 x 1024], K=512
    gemm_nt_wmma<EPI_SCALE_F32><<<dim3(SPAD / 64, SPAD / 256, BATCH), 256, 0, stream>>>(
        hb, hb, scores, nullptr, nullptr, DMODEL, DMODEL, DMODEL, SPAD, sscale,
        hStride, hStride, sStride);

    topk_softmax_kernel<<<dim3(SEQ, BATCH), 256, 0, stream>>>(scores, attn);

    // attn_out = attn · h_b  via NT with B = h_bᵀ, K = SPAD
    gemm_nt_wmma<EPI_F32><<<dim3(DMODEL / 64, SPAD / 256, BATCH), 256, 0, stream>>>(
        attn, hT, tmp, nullptr, nullptr, SPAD, SPAD, SPAD, DMODEL, 1.0f,
        sStride, tStride, hStride);

    add_ln_kernel<<<BATCH * SEQ, 256, 0, stream>>>(h, tmp, n1_s + l * DMODEL, n1_b + l * DMODEL, hb);

    // ---- FFN over all B*SPAD tokens at once (z = 1, strides 0) ----
    gemm_nt_wmma<EPI_BIAS_RELU_BF16><<<dim3(DIMFF / 64, (BATCH * SPAD) / 256, 1), 256, 0, stream>>>(
        hb, w1t, nullptr, mid, l1_b + l * DIMFF, DMODEL, DMODEL, DMODEL, DIMFF, 1.0f, 0, 0, 0);
    gemm_nt_wmma<EPI_BIAS_F32><<<dim3(DMODEL / 64, (BATCH * SPAD) / 256, 1), 256, 0, stream>>>(
        mid, w2t, tmp, nullptr, l2_b + l * DMODEL, DIMFF, DIMFF, DIMFF, DMODEL, 1.0f, 0, 0, 0);
    add_ln_kernel<<<BATCH * SEQ, 256, 0, stream>>>(h, tmp, n2_s + l * DMODEL, n2_b + l * DMODEL, hb);
  }

  decoder_kernel<<<1, 32, 0, stream>>>(h, dec_w, dec_b, out);
}
